// ResNet_cifar10_74242804678950
// MI455X (gfx1250) — 8-bit matrix operand order repaired and checked by reading, compiled, not run on hardware
//
#include <hip/hip_runtime.h>
#include <stdint.h>

typedef __attribute__((ext_vector_type(16))) int   v16i;
typedef __attribute__((ext_vector_type(8)))  float v8f;

#define EPSBN 1e-5f
#define FP8_P1 ((int8_t)0x38)
#define FP8_M1 ((int8_t)0xB8)

struct BCParams {
  const int8_t* inS;
  const int8_t* wpk;
  const float *bng, *bnb, *bnm, *bnv;
  const float* resid;
  float*  outF;
  int8_t* outS;
  double* pacc;
  int Hin, Win, Cpin;
  int Ho, Wo, Cout, Cpout;
  int stride, KW, tileOH, clip;
};

__global__ __launch_bounds__(256) void k_binconv(BCParams p) {
  __shared__ __align__(16) int8_t smem[43008];
  __shared__ double red[256];
  const int tid  = threadIdx.x;
  const int lane = tid & 31;
  const int wave = tid >> 5;
  const int hi   = lane >> 4;

  const int taps   = p.KW * p.KW;
  const int chunks = p.Cpin >> 7;
  const int padk   = (p.KW == 3) ? 1 : 0;
  const int TR     = (p.tileOH - 1) * p.stride + ((p.KW == 3) ? 3 : 1);
  const int TC     = (p.KW == 3) ? (p.Win + 2) : p.Win;

  const int tilesPerImg = p.Ho / p.tileOH;
  const int n    = blockIdx.x / tilesPerImg;
  const int ot   = blockIdx.x % tilesPerImg;
  const int oh0  = ot * p.tileOH;
  const int row0 = oh0 * p.stride - padk;

  {
    const int  cd    = p.Cpin >> 2;
    const int  total = TR * TC * cd;
    const int* src   = (const int*)p.inS;
    int*       dst   = (int*)smem;
    for (int i = tid; i < total; i += 256) {
      int c4 = i % cd;
      int rc = i / cd;
      int cc = rc % TC;
      int r  = rc / TC;
      int ih = row0 + r;
      int iw = cc - padk;
      int val0 = 0;
      if (ih >= 0 && ih < p.Hin && iw >= 0 && iw < p.Win)
        val0 = src[((size_t)(n * p.Hin + ih) * p.Win + iw) * cd + c4];
      dst[i] = val0;
    }
  }
  __syncthreads();

  double s1 = 0.0, s2 = 0.0;

  const int nMt = (p.tileOH * p.Wo) >> 4;
  const int nNt = p.Cout >> 4;
  const int nT  = nMt * nNt;
  const int m0  = lane & 15;

  for (int t = wave; t < nT; t += 8) {
    const int mtile = t % nMt;
    const int ntile = t / nMt;

    const int pA   = mtile * 16 + m0;
    const int ohlA = pA / p.Wo;
    const int owA  = pA - ohlA * p.Wo;

    v8f acc = {};
    for (int tap = 0; tap < taps; ++tap) {
      const int dy = tap / p.KW, dx = tap % p.KW;
      const int r  = ohlA * p.stride + dy;
      const int cc = owA * p.stride + dx;
      const int abase = ((r * TC + cc) * p.Cpin) >> 2;
      for (int kc = 0; kc < chunks; ++kc) {
        const int* ap = (const int*)smem + abase + (kc << 5);
        v16i a;
#pragma unroll
        for (int v = 0; v < 16; ++v) {
          const int kd = ((v >> 2) << 3) + (v & 3) + (hi << 2);
          a[v] = ap[kd];
        }
        const size_t wof = ((((size_t)(ntile * taps + tap)) * (size_t)chunks + kc) * 32 + lane) * 64;
        const int* bp = (const int*)(p.wpk + wof);
        __builtin_prefetch(bp + 512, 0, 1);
        v16i b;
#pragma unroll
        for (int v = 0; v < 16; ++v) b[v] = bp[v];
        acc = __builtin_amdgcn_wmma_f32_16x16x128_fp8_fp8(a, b, (short)0, acc, false, false);
      }
    }

    const int nch = ntile * 16 + (lane & 15);
#pragma unroll
    for (int r = 0; r < 8; ++r) {
      const int pM  = mtile * 16 + r + hi * 8;
      const int ohl = pM / p.Wo;
      const int ow  = pM - ohl * p.Wo;
      const int oh  = oh0 + ohl;
      const float raw = acc[r];
      s1 += (double)raw;
      s2 += (double)raw * (double)raw;
      float val = raw;
      const size_t oidx = (size_t)(n * p.Ho + oh) * p.Wo + ow;
      if (p.resid) val += p.resid[oidx * p.Cout + nch];
      if (p.bng) {
        const float sc = p.bng[nch] * rsqrtf(p.bnv[nch] + EPSBN);
        val = (val - p.bnm[nch]) * sc + p.bnb[nch];
      }
      if (p.clip) val = fminf(1.0f, fmaxf(-1.0f, val));
      if (p.outF) p.outF[oidx * p.Cout + nch] = val;
      if (p.outS) p.outS[oidx * p.Cpout + nch] = (val >= 0.0f) ? FP8_P1 : FP8_M1;
    }
  }

  red[tid] = s1;
  __syncthreads();
  for (int s = 128; s > 0; s >>= 1) { if (tid < s) red[tid] += red[tid + s]; __syncthreads(); }
  if (tid == 0) atomicAdd(&p.pacc[0], red[0]);
  __syncthreads();
  red[tid] = s2;
  __syncthreads();
  for (int s = 128; s > 0; s >>= 1) { if (tid < s) red[tid] += red[tid + s]; __syncthreads(); }
  if (tid == 0) atomicAdd(&p.pacc[1], red[0]);
}

__global__ void k_packw(const float* __restrict__ w, int8_t* __restrict__ dst,
                        int Cin, int taps, int chunks) {
  const size_t idx = (size_t)blockIdx.x * 256 + threadIdx.x;
  const int j    = (int)(idx & 63);
  const int lane = (int)((idx >> 6) & 31);
  size_t rest = idx >> 11;
  const int kc    = (int)(rest % chunks); rest /= chunks;
  const int tap   = (int)(rest % taps);
  const int ntile = (int)(rest / taps);
  const int v = j >> 2, bb = j & 3;
  const int K = ((v >> 2) << 5) + ((v & 3) << 2) + bb + ((lane >> 4) << 4);
  const int c = (kc << 7) + K;
  const int nch = ntile * 16 + (lane & 15);
  int8_t o = 0;
  if (c < Cin) {
    const float wv = w[((size_t)nch * Cin + c) * taps + tap];
    o = (wv >= 0.0f) ? FP8_P1 : FP8_M1;
  }
  dst[idx] = o;
}

__global__ __launch_bounds__(256) void k_conv1(const float* __restrict__ x,
                                               const float* __restrict__ w,
                                               const float* g, const float* b,
                                               const float* m, const float* vv,
                                               float* __restrict__ outF,
                                               int8_t* __restrict__ outS) {
  int idx = blockIdx.x * 256 + threadIdx.x;
  const int oc = idx % 80;  idx /= 80;
  const int ow = idx % 32;  idx /= 32;
  const int oh = idx % 32;
  const int n  = idx / 32;
  float acc = 0.f;
  for (int ci = 0; ci < 3; ++ci)
    for (int ky = 0; ky < 3; ++ky) {
      const int ih = oh + ky - 1;
      if (ih < 0 || ih >= 32) continue;
      for (int kx = 0; kx < 3; ++kx) {
        const int iw = ow + kx - 1;
        if (iw < 0 || iw >= 32) continue;
        acc += x[((size_t)(n * 3 + ci) * 32 + ih) * 32 + iw] *
               w[((oc * 3 + ci) * 3 + ky) * 3 + kx];
      }
    }
  const float sc  = g[oc] * rsqrtf(vv[oc] + EPSBN);
  float val = (acc - m[oc]) * sc + b[oc];
  val = fminf(1.0f, fmaxf(-1.0f, val));
  const size_t pix = (size_t)(n * 32 + oh) * 32 + ow;
  outF[pix * 80 + oc]  = val;
  outS[pix * 128 + oc] = (val >= 0.f) ? FP8_P1 : FP8_M1;
}

__global__ __launch_bounds__(320) void k_head(const float* __restrict__ h,
    const float* g2, const float* b2, const float* m2, const float* v2,
    const float* fw, const float* fb,
    const float* g3, const float* b3, const float* m3, const float* v3,
    float* __restrict__ out) {
  __shared__ float a[320];
  __shared__ float lg[10];
  const int n = blockIdx.x;
  const int c = threadIdx.x;
  if (c < 320) {
    float s = 0.f;
    for (int i = 0; i < 64; ++i) s += h[((size_t)n * 64 + i) * 320 + c];
    s *= (1.0f / 64.0f);
    s = (s - m2[c]) * (g2[c] * rsqrtf(v2[c] + EPSBN)) + b2[c];
    a[c] = fminf(1.0f, fmaxf(-1.0f, s));
  }
  __syncthreads();
  if (c < 10) {
    float t = fb[c];
    for (int i = 0; i < 320; ++i) t += a[i] * fw[c * 320 + i];
    lg[c] = (t - m3[c]) * (g3[c] * rsqrtf(v3[c] + EPSBN)) + b3[c];
  }
  __syncthreads();
  if (c == 0) {
    float mx = -1e30f;
    for (int o = 0; o < 10; ++o) mx = fmaxf(mx, lg[o]);
    float se = 0.f;
    for (int o = 0; o < 10; ++o) se += expf(lg[o] - mx);
    const float lse = logf(se);
    for (int o = 0; o < 10; ++o) out[n * 10 + o] = lg[o] - mx - lse;
  }
}

__global__ void k_psum_fin(const double* __restrict__ acc,
                           const float* __restrict__ psin,
                           float* __restrict__ out) {
  const int i = threadIdx.x;
  if (i >= 14) return;
  const float counts[14] = {
      20971520.f, 20971520.f, 20971520.f, 20971520.f,
      10485760.f, 10485760.f, 10485760.f, 10485760.f, 10485760.f,
      5242880.f,  5242880.f,  5242880.f,  5242880.f,  5242880.f};
  const double N  = (double)counts[i];
  const double S1 = acc[i * 2], S2 = acc[i * 2 + 1];
  const double mean2 = S1 / N;
  double var2 = (S2 - S1 * S1 / N) / (N - 1.0);
  if (var2 < 0.0) var2 = 0.0;
  const double sd2 = sqrt(var2);
  const double n1  = (double)psin[i * 3 + 0];
  const double m1  = (double)psin[i * 3 + 1];
  const double sd1 = (double)psin[i * 3 + 2];
  const double ntot = n1 + N;
  const double mtot = (m1 * n1 + mean2 * N) / ntot;
  const double left  = ((n1 - 1.0) * sd1 * sd1 + (N - 1.0) * sd2 * sd2) / (n1 + N - 1.0);
  const double right = n1 * N * (m1 - mean2) * (m1 - mean2) / ((n1 + N) * (n1 + N - 1.0));
  const double stot  = sqrt(left + right);
  out[2560 + i * 3 + 0] = (float)ntot;
  out[2560 + i * 3 + 1] = (float)mtot;
  out[2560 + i * 3 + 2] = (float)stot;
}

__global__ void k_zero128(int4* p) {
  p[(size_t)blockIdx.x * 256 + threadIdx.x] = make_int4(0, 0, 0, 0);
}
__global__ void k_zero_pacc(double* p) { if (threadIdx.x < 28) p[threadIdx.x] = 0.0; }

extern "C" void kernel_launch(void* const* d_in, const int* in_sizes, int n_in,
                              void* d_out, int out_size, void* d_ws, size_t ws_size,
                              hipStream_t stream) {
  (void)in_sizes; (void)n_in; (void)out_size; (void)ws_size;

  int8_t* ws = (int8_t*)d_ws;
  double* pacc    = (double*)ws;
  int8_t* wpkBase = ws + 1024;
  int8_t* SGA = ws + (16ull << 20);
  int8_t* SGB = ws + (48ull << 20);
  float*  FA  = (float*)(ws + (80ull  << 20));
  float*  FB  = (float*)(ws + (160ull << 20));
  float*  FC  = (float*)(ws + (240ull << 20));
  const size_t SIGN_BYTES = 32ull << 20;

  struct WL { int widx, Cout, Cin, taps, chunks; };
  const WL wl[14] = {
      {7,  80, 80, 9, 1},  {12, 80, 80, 9, 1},
      {17, 80, 80, 9, 1},  {22, 80, 80, 9, 1},
      {27, 160, 80, 9, 1}, {32, 160, 160, 9, 2},
      {37, 160, 80, 1, 1},
      {42, 160, 160, 9, 2}, {47, 160, 160, 9, 2},
      {52, 320, 160, 9, 2}, {57, 320, 320, 9, 3},
      {62, 320, 160, 1, 2},
      {67, 320, 320, 9, 3}, {72, 320, 320, 9, 3}};
  int8_t* wpk[14];
  size_t woff = 0;
  for (int i = 0; i < 14; ++i) {
    wpk[i] = wpkBase + woff;
    const size_t bytes = (size_t)(wl[i].Cout / 16) * wl[i].taps * wl[i].chunks * 2048;
    woff += bytes;
    k_packw<<<(unsigned)(bytes / 256), 256, 0, stream>>>(
        (const float*)d_in[wl[i].widx], wpk[i], wl[i].Cin, wl[i].taps, wl[i].chunks);
  }

  k_zero_pacc<<<1, 32, 0, stream>>>(pacc);
  auto zero8 = [&](int8_t* ptr) {
    k_zero128<<<(unsigned)(SIGN_BYTES / (16 * 256)), 256, 0, stream>>>((int4*)ptr);
  };
  zero8(SGA);
  zero8(SGB);

  k_conv1<<<81920, 256, 0, stream>>>(
      (const float*)d_in[0], (const float*)d_in[2],
      (const float*)d_in[3], (const float*)d_in[4],
      (const float*)d_in[5], (const float*)d_in[6], FA, SGA);

  auto bc = [&](const int8_t* inS, const int8_t* wp, int bnBase, bool clip,
                const float* resid, float* outF, int8_t* outS, int pidx,
                int Hin, int Win, int Cpin, int Ho, int Wo, int Cout, int Cpout,
                int stride, int KW, int tileOH) {
    BCParams q{};
    q.inS = inS; q.wpk = wp;
    if (bnBase >= 0) {
      q.bng = (const float*)d_in[bnBase + 0];
      q.bnb = (const float*)d_in[bnBase + 1];
      q.bnm = (const float*)d_in[bnBase + 2];
      q.bnv = (const float*)d_in[bnBase + 3];
    }
    q.resid = resid; q.outF = outF; q.outS = outS;
    q.pacc = pacc + pidx * 2;
    q.Hin = Hin; q.Win = Win; q.Cpin = Cpin;
    q.Ho = Ho; q.Wo = Wo; q.Cout = Cout; q.Cpout = Cpout;
    q.stride = stride; q.KW = KW; q.tileOH = tileOH; q.clip = clip ? 1 : 0;
    k_binconv<<<256 * (Ho / tileOH), 256, 0, stream>>>(q);
  };

  bc(SGA, wpk[0], 8,  true, nullptr, nullptr, SGB, 0, 32,32,128, 32,32,80,128, 1,3,4);
  bc(SGB, wpk[1], 13, true, FA,      FB,      SGA, 1, 32,32,128, 32,32,80,128, 1,3,4);
  bc(SGA, wpk[2], 18, true, nullptr, nullptr, SGB, 2, 32,32,128, 32,32,80,128, 1,3,4);
  bc(SGB, wpk[3], 23, true, FB,      FA,      SGA, 3, 32,32,128, 32,32,80,128, 1,3,4);
  bc(SGA, wpk[6], 38, false, nullptr, FC, nullptr, 6, 32,32,128, 16,16,160,0, 2,1,4);
  zero8(SGB);
  bc(SGA, wpk[4], 28, true, nullptr, nullptr, SGB, 4, 32,32,128, 16,16,160,256, 2,3,4);
  zero8(SGA);
  bc(SGB, wpk[5], 33, true, FC, FB, SGA, 5, 16,16,256, 16,16,160,256, 1,3,4);
  bc(SGA, wpk[7], 43, true, nullptr, nullptr, SGB, 7, 16,16,256, 16,16,160,256, 1,3,4);
  bc(SGB, wpk[8], 48, true, FB,      FA,      SGA, 8, 16,16,256, 16,16,160,256, 1,3,4);
  bc(SGA, wpk[11], 63, false, nullptr, FC, nullptr, 11, 16,16,256, 8,8,320,0, 2,1,4);
  zero8(SGB);
  bc(SGA, wpk[9], 53, true, nullptr, nullptr, SGB, 9, 16,16,256, 8,8,320,384, 2,3,4);
  zero8(SGA);
  bc(SGB, wpk[10], 58, true, FC, FB, SGA, 10, 8,8,384, 8,8,320,384, 1,3,8);
  bc(SGA, wpk[12], 68, true, nullptr, nullptr, SGB, 12, 8,8,384, 8,8,320,384, 1,3,8);
  bc(SGB, wpk[13], -1, false, FB, FA, nullptr, 13, 8,8,384, 8,8,320,384, 1,3,8);

  k_head<<<256, 320, 0, stream>>>(
      FA,
      (const float*)d_in[77], (const float*)d_in[78], (const float*)d_in[79], (const float*)d_in[80],
      (const float*)d_in[85], (const float*)d_in[86],
      (const float*)d_in[81], (const float*)d_in[82], (const float*)d_in[83], (const float*)d_in[84],
      (float*)d_out);
  k_psum_fin<<<1, 32, 0, stream>>>(pacc, (const float*)d_in[1], (float*)d_out);
}
